// CausalSelfAttention_75986561401490
// MI455X (gfx1250) — matrix operand order repaired and checked by reading, not compiled, not run on hardware
//
#include <hip/hip_runtime.h>

typedef __attribute__((ext_vector_type(16))) _Float16 v16h;
typedef __attribute__((ext_vector_type(8)))  _Float16 v8h;
typedef __attribute__((ext_vector_type(4)))  _Float16 v4h;
typedef __attribute__((ext_vector_type(8)))  float    v8f;

#define WMMA16(A, B, C) __builtin_amdgcn_wmma_f32_16x16x32_f16( \
    false, (A), false, (B), (short)0, (C), false, false)

__device__ __forceinline__ v16h cat8(v8h lo, v8h hi) {
  return __builtin_shufflevector(lo, hi, 0,1,2,3,4,5,6,7,8,9,10,11,12,13,14,15);
}
// A-fragment: 16 contiguous K halves per lane
__device__ __forceinline__ v16h ld_afrag(const _Float16* p) {
  v8h a = *(const v8h*)p;
  v8h b = *(const v8h*)(p + 8);
  return cat8(a, b);
}
// B-fragment: 16 contiguous K halves per lane
__device__ __forceinline__ v16h ld_bfrag(const _Float16* p) {
  v8h a = *(const v8h*)p;
  v8h b = *(const v8h*)(p + 8);
  return cat8(a, b);
}
// Async global -> LDS copy, 16B per lane, tracked by ASYNCcnt (ISA 15.18.3 op 98)
__device__ __forceinline__ void async_ld16(unsigned lds_off, const _Float16* g) {
  asm volatile("global_load_async_to_lds_b128 %0, %1, off"
               :: "v"(lds_off), "v"(g) : "memory");
}

// ---------------------------------------------------------------------------
// Kernel 1: x fp32 -> f16 (same layout), 4 elems/thread
// ---------------------------------------------------------------------------
__global__ __launch_bounds__(256) void xconv_kernel(const float* __restrict__ x,
                                                    _Float16* __restrict__ xh) {
  const size_t idx = (size_t)blockIdx.x * 256 + threadIdx.x;  // float4 index
  float4 f = ((const float4*)x)[idx];
  v4h h = {(_Float16)f.x, (_Float16)f.y, (_Float16)f.z, (_Float16)f.w};
  ((v4h*)xh)[idx] = h;
}

// ---------------------------------------------------------------------------
// Kernel 2: weight transpose + fp32 -> f16:  w[K][N] -> wt[N][K]
// ---------------------------------------------------------------------------
__global__ void wtrans_kernel(const float* __restrict__ w, _Float16* __restrict__ wt,
                              int K, int N) {
  long idx = (long)blockIdx.x * 256 + threadIdx.x;
  if (idx >= (long)K * N) return;
  int k = (int)(idx / N);
  int n = (int)(idx % N);
  wt[(size_t)n * K + k] = (_Float16)w[idx];
}

// ---------------------------------------------------------------------------
// Kernel 3: unified f16 GEMM  a[8192][2048] @ wt[N][2048]^T + bias.
// Block tile 128x128 (8 waves, each 64x32 -> 8 WMMA/k-step).
// Double-buffered LDS fed by global_load_async_to_lds_b128 (ASYNCcnt).
// MODE 0: qkv epilogue (scatter q/k/v^T, f16, 1/sqrt(D) folded into q).
// MODE 1: proj epilogue (fp32 out + bias).
// ---------------------------------------------------------------------------
template <int MODE>
__global__ __launch_bounds__(256) void gemm_f16_kernel(
    const _Float16* __restrict__ a, const _Float16* __restrict__ wt,
    const float* __restrict__ bias,
    _Float16* __restrict__ qh, _Float16* __restrict__ kh, _Float16* __restrict__ vth,
    float* __restrict__ out) {
  __shared__ _Float16 As[2][128][40];   // 10240 B per buffer, rows padded to 80 B
  __shared__ _Float16 Bs[2][128][40];
  const int tid = threadIdx.x;
  const int wid = tid >> 5, lane = tid & 31;
  const int hl = lane >> 4, lm = lane & 15;
  const int m0 = (wid & 1) * 64, n0 = (wid >> 1) * 32;
  const size_t mBase = (size_t)blockIdx.x * 128;
  const size_t nBase = (size_t)blockIdx.y * 128;
  const int ld_row = tid >> 1, ld_seg = tid & 1;  // 128 rows x 2 x 32B

  const _Float16* agp = a  + (mBase + ld_row) * 2048 + ld_seg * 16;
  const _Float16* bgp = wt + (nBase + ld_row) * 2048 + ld_seg * 16;
  // addrspace(3) pointers: low 32 bits of the flat address are the LDS offset
  const unsigned aL0 = (unsigned)(size_t)&As[0][ld_row][ld_seg * 16];
  const unsigned bL0 = (unsigned)(size_t)&Bs[0][ld_row][ld_seg * 16];
  const unsigned BUFB = 128u * 40u * 2u;  // 10240 bytes between buffers

  // prologue: async-load tile 0 into buffer 0
  async_ld16(aL0,      agp);
  async_ld16(aL0 + 16, agp + 8);
  async_ld16(bL0,      bgp);
  async_ld16(bL0 + 16, bgp + 8);

  v8f acc[4][2] = {};

  for (int k0 = 0; k0 < 2048; k0 += 32) {
    const int cur = (k0 >> 5) & 1;
    if (k0 + 32 < 2048) {  // prefetch next tile into the other buffer
      const unsigned nb = (1 - cur) * BUFB;
      const _Float16* ag = agp + k0 + 32;
      const _Float16* bg = bgp + k0 + 32;
      async_ld16(aL0 + nb,      ag);
      async_ld16(aL0 + nb + 16, ag + 8);
      async_ld16(bL0 + nb,      bg);
      async_ld16(bL0 + nb + 16, bg + 8);
      asm volatile("s_wait_asynccnt 0x4" ::: "memory");  // tile k0 done (in-order)
    } else {
      asm volatile("s_wait_asynccnt 0x0" ::: "memory");
    }
    __syncthreads();  // tile k0 visible from every wave

    v16h af[4], bf[2];
#pragma unroll
    for (int mi = 0; mi < 4; ++mi)
      af[mi] = ld_afrag(&As[cur][m0 + mi * 16 + lm][hl * 16]);
#pragma unroll
    for (int ni = 0; ni < 2; ++ni)
      bf[ni] = ld_bfrag(&Bs[cur][n0 + ni * 16 + lm][hl * 16]);
#pragma unroll
    for (int mi = 0; mi < 4; ++mi)
#pragma unroll
      for (int ni = 0; ni < 2; ++ni)
        acc[mi][ni] = WMMA16(af[mi], bf[ni], acc[mi][ni]);

    __syncthreads();  // everyone done reading buf cur before it is re-filled
  }

#pragma unroll
  for (int ni = 0; ni < 2; ++ni) {
    const int ng = (int)nBase + n0 + ni * 16 + lm;
    const float bv = bias[ng];
    if (MODE == 0) {
      const int which = ng >> 11;  // 0=q 1=k 2=v
      const int r2 = ng & 2047;
      const int hh = r2 >> 7, dd = r2 & 127;
#pragma unroll
      for (int mi = 0; mi < 4; ++mi)
#pragma unroll
        for (int r = 0; r < 8; ++r) {
          const int tm = (int)mBase + m0 + mi * 16 + r + 8 * hl;
          const int bb = tm >> 11, tt = tm & 2047;
          float val = acc[mi][ni][r] + bv;
          if (which == 0) {
            val *= 0.08838834764831845f;  // 1/sqrt(128)
            qh[((size_t)(bb * 16 + hh) * 2048 + tt) * 128 + dd] = (_Float16)val;
          } else if (which == 1) {
            kh[((size_t)(bb * 16 + hh) * 2048 + tt) * 128 + dd] = (_Float16)val;
          } else {
            vth[((size_t)(bb * 16 + hh) * 128 + dd) * 2048 + tt] = (_Float16)val;
          }
        }
    } else {
#pragma unroll
      for (int mi = 0; mi < 4; ++mi)
#pragma unroll
        for (int r = 0; r < 8; ++r) {
          const size_t tm = mBase + m0 + mi * 16 + r + 8 * hl;
          out[tm * 2048 + ng] = acc[mi][ni][r] + bv;
        }
    }
  }
}

// ---------------------------------------------------------------------------
// Kernel 4: flash attention, one wave per 16-row Q tile, 64-key tiles,
// online softmax, P routed through per-wave LDS to become A-fragments.
// ---------------------------------------------------------------------------
__global__ __launch_bounds__(256) void attn_kernel(
    const _Float16* __restrict__ qh, const _Float16* __restrict__ kh,
    const _Float16* __restrict__ vth, _Float16* __restrict__ oh) {
  __shared__ _Float16 Pb[8][16 * 72];  // per-wave 16x64 P tile, rows padded to 144 B
  const int tid = threadIdx.x;
  const int wid = tid >> 5, lane = tid & 31;
  const int hl = lane >> 4, lm = lane & 15;
  const int gw = blockIdx.x * 8 + wid;  // 8192 q-tiles; all waves of a block share (b,h)
  const int b  = gw >> 11;
  const int hh = (gw >> 7) & 15;
  const int qt = gw & 127;
  const int qBase = qt << 4;

  const _Float16* Qp = qh  + ((size_t)(b * 16 + hh) * 2048) * 128;
  const _Float16* Kp = kh  + ((size_t)(b * 16 + hh) * 2048) * 128;
  const _Float16* Vp = vth + ((size_t)(b * 16 + hh) * 128) * 2048;
  _Float16* Pw = &Pb[wid][0];

  v16h qf[4];
#pragma unroll
  for (int dc = 0; dc < 4; ++dc)
    qf[dc] = ld_afrag(Qp + (size_t)(qBase + lm) * 128 + dc * 32 + hl * 16);

  v8f o[8] = {};
  float mx[8], sm[8];
#pragma unroll
  for (int r = 0; r < 8; ++r) { mx[r] = -3.0e38f; sm[r] = 0.0f; }

  const int nkt = (qBase + 15) >> 6;  // last causal 64-key tile
  for (int kt = 0; kt <= nkt; ++kt) {
    const int kb = kt << 6;
    // S = Q @ K^T : four 16x16 key sub-tiles
    v8f s[4] = {};
#pragma unroll
    for (int dc = 0; dc < 4; ++dc) {
      const _Float16* kbase = Kp + (size_t)kb * 128 + dc * 32 + hl * 16;
#pragma unroll
      for (int j = 0; j < 4; ++j) {
        v16h kf = ld_bfrag(kbase + (size_t)(j * 16 + lm) * 128);
        s[j] = WMMA16(qf[dc], kf, s[j]);
      }
    }
    // causal mask + online softmax over 64 keys
#pragma unroll
    for (int r = 0; r < 8; ++r) {
      const int qr = qBase + r + 8 * hl;
      float pv[4];
#pragma unroll
      for (int j = 0; j < 4; ++j)
        pv[j] = (kb + j * 16 + lm <= qr) ? s[j][r] : -3.0e38f;
      float vmax = fmaxf(fmaxf(pv[0], pv[1]), fmaxf(pv[2], pv[3]));
#pragma unroll
      for (int off = 8; off > 0; off >>= 1)
        vmax = fmaxf(vmax, __shfl_xor(vmax, off, 32));
      const float mnew = fmaxf(mx[r], vmax);
      const float alpha = __expf(mx[r] - mnew);
      float ps = 0.0f;
#pragma unroll
      for (int j = 0; j < 4; ++j) {
        pv[j] = __expf(pv[j] - mnew);
        ps += pv[j];
      }
#pragma unroll
      for (int off = 8; off > 0; off >>= 1)
        ps += __shfl_xor(ps, off, 32);
      sm[r] = sm[r] * alpha + ps;
      mx[r] = mnew;
#pragma unroll
      for (int dt = 0; dt < 8; ++dt) o[dt][r] *= alpha;
#pragma unroll
      for (int j = 0; j < 4; ++j)
        Pw[(r + 8 * hl) * 72 + j * 16 + lm] = (_Float16)pv[j];
    }
    asm volatile("s_wait_dscnt 0x0" ::: "memory");  // intra-wave LDS RAW
    v16h pf0 = ld_afrag(Pw + lm * 72 + hl * 16);        // keys kb+0..31
    v16h pf1 = ld_afrag(Pw + lm * 72 + 32 + hl * 16);   // keys kb+32..63
    // O += P @ V  (V^T rows -> contiguous B-fragments)
#pragma unroll
    for (int dt = 0; dt < 8; ++dt) {
      const _Float16* vb = Vp + (size_t)(dt * 16 + lm) * 2048 + kb + hl * 16;
      o[dt] = WMMA16(pf0, ld_bfrag(vb), o[dt]);
      o[dt] = WMMA16(pf1, ld_bfrag(vb + 32), o[dt]);
    }
  }

  // normalize + store [B*T][H*D] f16 for the projection GEMM
#pragma unroll
  for (int r = 0; r < 8; ++r) {
    const float inv = 1.0f / sm[r];
    const size_t rowoff =
        ((size_t)(b * 2048 + qBase + r + 8 * hl)) * 2048 + hh * 128;
#pragma unroll
    for (int dt = 0; dt < 8; ++dt)
      oh[rowoff + dt * 16 + lm] = (_Float16)(o[dt][r] * inv);
  }
}

// ---------------------------------------------------------------------------
// Launch.  Workspace layout (halfs), ~201 MB total:
//   wqkv_t[6144][2048]  wproj_t[2048][2048]  q/k[B,H,T,D]  v^T[B,H,D,T]
//   attn[B*T][H*D]      x_h[B*T][C]
// ---------------------------------------------------------------------------
extern "C" void kernel_launch(void* const* d_in, const int* in_sizes, int n_in,
                              void* d_out, int out_size, void* d_ws, size_t ws_size,
                              hipStream_t stream) {
  const float* x      = (const float*)d_in[0];
  const float* w_qkv  = (const float*)d_in[1];
  const float* b_qkv  = (const float*)d_in[2];
  const float* w_proj = (const float*)d_in[3];
  const float* b_proj = (const float*)d_in[4];
  float* out = (float*)d_out;

  _Float16* ws      = (_Float16*)d_ws;
  _Float16* wqkv_t  = ws;
  _Float16* wproj_t = wqkv_t + (size_t)6144 * 2048;
  _Float16* q_h     = wproj_t + (size_t)2048 * 2048;
  _Float16* k_h     = q_h  + (size_t)8192 * 2048;
  _Float16* vt_h    = k_h  + (size_t)8192 * 2048;
  _Float16* attn_h  = vt_h + (size_t)8192 * 2048;
  _Float16* x_h     = attn_h + (size_t)8192 * 2048;

  xconv_kernel<<<dim3(16384), dim3(256), 0, stream>>>(x, x_h);
  wtrans_kernel<<<dim3((2048 * 6144) / 256), dim3(256), 0, stream>>>(
      w_qkv, wqkv_t, 2048, 6144);
  wtrans_kernel<<<dim3((2048 * 2048) / 256), dim3(256), 0, stream>>>(
      w_proj, wproj_t, 2048, 2048);
  gemm_f16_kernel<0><<<dim3(64, 48), dim3(256), 0, stream>>>(
      x_h, wqkv_t, b_qkv, q_h, k_h, vt_h, nullptr);
  attn_kernel<<<dim3(1024), dim3(256), 0, stream>>>(q_h, k_h, vt_h, attn_h);
  gemm_f16_kernel<1><<<dim3(64, 16), dim3(256), 0, stream>>>(
      attn_h, wproj_t, b_proj, nullptr, nullptr, nullptr, out);
}
